// AtomEncoder_317827580848
// MI455X (gfx1250) — matrix operand order repaired and checked by reading, not compiled, not run on hardware
//
#include <hip/hip_runtime.h>

// ---------------------------------------------------------------------------
// AtomEncoder for MI455X (gfx1250, wave32, WMMA 16x16x32 f16 + TDM staging)
// LDS tiles use bank-conflict-free leading dims (stride % 64 banks != 0).
// ---------------------------------------------------------------------------

typedef __attribute__((ext_vector_type(16))) _Float16     v16h;
typedef __attribute__((ext_vector_type(8)))  float        v8f;
typedef __attribute__((ext_vector_type(4)))  unsigned int v4u;
typedef __attribute__((ext_vector_type(8)))  int          v8i;
typedef __attribute__((ext_vector_type(4)))  int          v4i;

#define N_ATOMS 4096
#define N_TOK   512
#define ATOM_S  128
#define ATOM_Z  16
#define TOKEN_S 384
#define TOKEN_Z 128
#define FEATN   388

#define LDZ 132   // k_zp LDS stride  (132 % 64 = 4  -> conflict-free)
#define LDS2 388  // k_s2c LDS stride (388 % 64 = 4)
#define LDE 420   // k_embed LDS stride (420 % 64 = 36)
#define LDP 17    // k_p 16x16 tile stride (17 coprime with 64)

__device__ __forceinline__ v8f wmma_f16(v16h a, v16h b, v8f c) {
  // (neg_a, A, neg_b, B, c_mod, C, reuse_a, reuse_b)
  return __builtin_amdgcn_wmma_f32_16x16x32_f16(false, a, false, b, (short)0, c,
                                                false, false);
}

// A fragment (16x32, row-major f32 tile in LDS, leading dim ld).
// lane L: m=L&15, kb=(L>>4)*8; elems[0..7]=A[m][kb+j], [8..15]=A[m][kb+16+j].
__device__ __forceinline__ v16h afrag(const float* t, int ld, int lane) {
  int m = lane & 15, kb = (lane >> 4) * 8;
  const float* r = t + m * ld + kb;
  v16h a;
#pragma unroll
  for (int j = 0; j < 8; ++j) {
    a[j]     = (_Float16)r[j];
    a[j + 8] = (_Float16)r[16 + j];
  }
  return a;
}

// A fragment from a relu'd 16x16 LDS tile, stride LDP (K padded 16 -> 32).
__device__ __forceinline__ v16h afrag16_relu(const float* t, int lane) {
  int m = lane & 15, kb = (lane >> 4) * 8;
  v16h a;
#pragma unroll
  for (int j = 0; j < 8; ++j) {
    a[j]     = (_Float16)fmaxf(t[m * LDP + kb + j], 0.f);
    a[j + 8] = (_Float16)0.f;
  }
  return a;
}

// Pre-swizzled B fragment: 32 lanes x 16 halves, contiguous -> 2x b128 loads.
__device__ __forceinline__ v16h bload(const _Float16* base, int frag, int lane) {
  return *(const v16h*)(base + ((frag << 5) + lane) * 16);
}

// ---------------------------------------------------------------------------
// Prep: swizzle a [K, ncols] f32 weight matrix into B-fragment order (f16).
// dst[((kc*ntiles+t)*32+lane)*16+j] = W[kc*32+(lane>>4)*8+(j&7)+(j>>3)*16][t*16+(lane&15)]
// zero-padded past K. One thread per output half.
// ---------------------------------------------------------------------------
__global__ __launch_bounds__(256) void k_swzB(const float* __restrict__ src,
                                              int K, int ncols, int nfrag,
                                              int ntiles,
                                              _Float16* __restrict__ dst) {
  int idx = blockIdx.x * 256 + threadIdx.x;
  if (idx >= nfrag * 512) return;
  int frag = idx >> 9, w = idx & 511;
  int lane = w >> 4, j = w & 15;
  int kc = frag / ntiles, t = frag - kc * ntiles;
  int n = t * 16 + (lane & 15);
  int k = kc * 32 + ((lane >> 4) * 8) + (j & 7) + ((j >> 3) * 16);
  dst[idx] = (k < K) ? (_Float16)src[k * ncols + n] : (_Float16)0.f;
}

// ---------------------------------------------------------------------------
// Kernel 1: c0 = atom_feats @ W_feat + b_feat   (also q = c0)
// block 256 (8 waves) = 16 atoms x 128 cols; grid 256.
// ---------------------------------------------------------------------------
__global__ __launch_bounds__(256) void k_embed(
    const float* __restrict__ pos, const float* __restrict__ charge,
    const float* __restrict__ elem, const float* __restrict__ chars,
    const _Float16* __restrict__ hWf, const float* __restrict__ bfeat,
    float* __restrict__ c0, float* __restrict__ q_out) {
  __shared__ float At[16 * LDE];  // 388 padded to 416 (13 K-chunks), ld 420
  int tid = threadIdx.x;
  int n0 = blockIdx.x * 16;
  for (int idx = tid; idx < 16 * LDE; idx += 256) {
    int r = idx / LDE, f = idx - r * LDE;
    int n = n0 + r;
    float v = 0.f;
    if (f < 3)        v = pos[n * 3 + f];
    else if (f == 3)  v = charge[n];
    else if (f < 132) v = elem[n * 128 + (f - 4)];
    else if (f < 388) v = chars[n * 256 + (f - 132)];
    At[idx] = v;
  }
  __syncthreads();

  int wave = tid >> 5, lane = tid & 31;
  v8f acc = {};
#pragma unroll 2
  for (int kc = 0; kc < 13; ++kc) {
    v16h a = afrag(&At[kc * 32], LDE, lane);
    v16h b = bload(hWf, kc * 8 + wave, lane);
    acc = wmma_f16(a, b, acc);
  }
#pragma unroll
  for (int r = 0; r < 8; ++r) {
    int m   = (lane >> 4) * 8 + r;
    int col = wave * 16 + (lane & 15);
    float v = acc[r] + bfeat[col];
    int off = (n0 + m) * ATOM_S + col;
    c0[off]    = v;
    q_out[off] = v;
  }
}

// ---------------------------------------------------------------------------
// Kernel 2: s2c = LN(s_trunk) @ W_s2c   ([512,384] -> [512,128]); grid 32.
// ---------------------------------------------------------------------------
__global__ __launch_bounds__(256) void k_s2c(
    const float* __restrict__ s_trunk, const float* __restrict__ g,
    const float* __restrict__ bln, const _Float16* __restrict__ hW,
    float* __restrict__ s2c) {
  __shared__ float At[16 * LDS2];
  int tid = threadIdx.x;
  int r0 = blockIdx.x * 16;
  for (int idx = tid; idx < 16 * TOKEN_S; idx += 256) {
    int row = idx / TOKEN_S, col = idx - row * TOKEN_S;
    At[row * LDS2 + col] = s_trunk[r0 * TOKEN_S + idx];
  }
  __syncthreads();

  // LN: 16 threads per row, shuffle-reduced within aligned 16-lane groups
  {
    int row = tid >> 4, part = tid & 15;
    float s = 0.f, ss = 0.f;
    for (int f = part; f < TOKEN_S; f += 16) {
      float x = At[row * LDS2 + f];
      s += x; ss += x * x;
    }
#pragma unroll
    for (int off = 8; off; off >>= 1) {
      s  += __shfl_xor(s, off, 32);
      ss += __shfl_xor(ss, off, 32);
    }
    float m  = s * (1.f / TOKEN_S);
    float v  = ss * (1.f / TOKEN_S) - m * m;
    float rs = rsqrtf(v + 1e-5f);
    for (int f = part; f < TOKEN_S; f += 16)
      At[row * LDS2 + f] = (At[row * LDS2 + f] - m) * rs * g[f] + bln[f];
  }
  __syncthreads();

  int wave = tid >> 5, lane = tid & 31;
  v8f acc = {};
#pragma unroll 2
  for (int kc = 0; kc < 12; ++kc) {
    v16h a = afrag(&At[kc * 32], LDS2, lane);
    v16h b = bload(hW, kc * 8 + wave, lane);
    acc = wmma_f16(a, b, acc);
  }
#pragma unroll
  for (int r = 0; r < 8; ++r) {
    int m   = (lane >> 4) * 8 + r;
    int col = wave * 16 + (lane & 15);
    s2c[(r0 + m) * ATOM_S + col] = acc[r];
  }
}

// ---------------------------------------------------------------------------
// Kernel 3 (bandwidth-dominant: streams all 134 MB of z once):
// zp = LN(z) @ W_z2p, 262144 rows of 128 -> 16.
// z tile (64 rows x 128 f32) staged to LDS by the Tensor Data Mover, which
// also inserts a 4-DWORD pad every 128 DWORDs (ld 132 -> bank-conflict-free).
// block 128 (4 waves); grid 4096.
// ---------------------------------------------------------------------------
__global__ __launch_bounds__(128) void k_zp(
    const float* __restrict__ z, const float* __restrict__ g,
    const float* __restrict__ bln, const _Float16* __restrict__ hW,
    float* __restrict__ zp) {
  __shared__ float Zt[64 * LDZ];  // 33 KB, rows padded by TDM
  __shared__ float Gs[TOKEN_Z], Bs[TOKEN_Z];
  int tid = threadIdx.x, wave = tid >> 5, lane = tid & 31;
  long r0 = (long)blockIdx.x * 64;

  if (tid < TOKEN_Z) { Gs[tid] = g[tid]; Bs[tid] = bln[tid]; }

  if (wave == 0) {
    // Tensor DMA descriptor (D#): 2D tile 128 x 64 elems, 4B each,
    // tensor_dim0_stride = 128; LDS pad: every 128 DWORDs add 4 DWORDs.
    unsigned long long ga = (unsigned long long)(z + r0 * TOKEN_Z);
    unsigned int lds = (unsigned int)(unsigned long long)&Zt[0];
    v4u g0;
    g0[0] = 1u;                                   // count=1 (valid, user mode)
    g0[1] = lds;                                  // lds_addr (bytes)
    g0[2] = (unsigned int)(ga & 0xFFFFFFFFu);     // global_addr[31:0]
    g0[3] = (unsigned int)((ga >> 32) & 0x1FFFFFFu) | (2u << 30);  // type=2
    v8i g1;
    // wg_mask=0 | data_size=2 (4B) | pad_enable | pad_interval=6 (128 dw)
    // | pad_amount=3 (4 dw)
    g1[0] = (int)((2u << 16) | (1u << 20) | (6u << 22) | (3u << 25));
    g1[1] = (int)(128u << 16);          // tensor_dim0 = 128 (bits 79:48)
    g1[2] = 0;                          // dim0 hi / dim1 lo
    g1[3] = (int)(4u | (128u << 16));   // tensor_dim1 = 0x40000, tile_dim0=128
    g1[4] = 64;                         // tile_dim1=64, tile_dim2=0
    g1[5] = 128;                        // tensor_dim0_stride = 128
    g1[6] = 0;
    g1[7] = 0;
    v4i gz = {0, 0, 0, 0};
#if __clang_major__ >= 23
    v8i gz8 = {0, 0, 0, 0, 0, 0, 0, 0};
    __builtin_amdgcn_tensor_load_to_lds(g0, g1, gz, gz, gz8, 0);
#else
    __builtin_amdgcn_tensor_load_to_lds(g0, g1, gz, gz, 0);
#endif
    __builtin_amdgcn_s_wait_tensorcnt(0);
  }
  __syncthreads();

  // LN: one thread-pair per row; partner combine via shfl_xor(1)
  {
    int row = tid >> 1, half = tid & 1;
    float* zr = &Zt[row * LDZ + half * 64];
    float s = 0.f, ss = 0.f;
#pragma unroll 4
    for (int f = 0; f < 64; ++f) { float x = zr[f]; s += x; ss += x * x; }
    s  += __shfl_xor(s, 1, 32);
    ss += __shfl_xor(ss, 1, 32);
    float m  = s * (1.f / TOKEN_Z);
    float v  = ss * (1.f / TOKEN_Z) - m * m;
    float rs = rsqrtf(v + 1e-5f);
    const float* gg = &Gs[half * 64];
    const float* bb = &Bs[half * 64];
#pragma unroll 4
    for (int f = 0; f < 64; ++f) zr[f] = (zr[f] - m) * rs * gg[f] + bb[f];
  }
  __syncthreads();

  const float* tile = &Zt[wave * 16 * LDZ];
  v8f acc = {};
#pragma unroll
  for (int kc = 0; kc < 4; ++kc)
    acc = wmma_f16(afrag(tile + kc * 32, LDZ, lane), bload(hW, kc, lane), acc);
#pragma unroll
  for (int r = 0; r < 8; ++r) {
    long row = r0 + wave * 16 + (lane >> 4) * 8 + r;
    int  col = lane & 15;
    zp[row * ATOM_Z + col] = acc[r];
  }
}

// ---------------------------------------------------------------------------
// Kernel 4: token index from one-hot (wave min-reduce) + c = c0 + s2c[tok]
// one wave per atom; block 256 (8 atoms); grid 512.
// ---------------------------------------------------------------------------
__global__ __launch_bounds__(256) void k_token(
    const float* __restrict__ a2t, const float* __restrict__ c0,
    const float* __restrict__ s2c, float* __restrict__ c1,
    float* __restrict__ c_out, int* __restrict__ tok) {
  int tid = threadIdx.x, wave = tid >> 5, lane = tid & 31;
  int n = blockIdx.x * 8 + wave;
  int found = 1 << 30;
  const float* row = a2t + (long)n * N_TOK + lane * 16;
  for (int j = 0; j < 16; ++j)
    if (row[j] > 0.5f) { found = lane * 16 + j; break; }
#pragma unroll
  for (int off = 16; off > 0; off >>= 1)
    found = min(found, __shfl_xor(found, off, 32));
  if (found == (1 << 30)) found = 0;
  if (lane == 0) tok[n] = found;
  for (int col = lane; col < ATOM_S; col += 32) {
    float v = c0[n * ATOM_S + col] + s2c[found * ATOM_S + col];
    c1[n * ATOM_S + col]    = v;
    c_out[n * ATOM_S + col] = v;
  }
}

// ---------------------------------------------------------------------------
// Kernel 5: cqp = relu(c) @ W_cq, ckp = relu(c) @ W_ck  (per atom, 128 -> 16)
// block 256 (8 waves x 16 atoms); grid 32.
// ---------------------------------------------------------------------------
__global__ __launch_bounds__(256) void k_proj(
    const float* __restrict__ c1, const _Float16* __restrict__ hWcq,
    const _Float16* __restrict__ hWck, float* __restrict__ cqp,
    float* __restrict__ ckp) {
  int tid = threadIdx.x, wave = tid >> 5, lane = tid & 31;
  int n0 = blockIdx.x * 128 + wave * 16;
  v8f aq = {}, ak = {};
#pragma unroll
  for (int kc = 0; kc < 4; ++kc) {
    int m = lane & 15, kb = kc * 32 + (lane >> 4) * 8;
    const float* r = c1 + (n0 + m) * ATOM_S + kb;
    v16h a;
#pragma unroll
    for (int j = 0; j < 8; ++j) {
      a[j]     = (_Float16)fmaxf(r[j], 0.f);
      a[j + 8] = (_Float16)fmaxf(r[16 + j], 0.f);
    }
    aq = wmma_f16(a, bload(hWcq, kc, lane), aq);
    ak = wmma_f16(a, bload(hWck, kc, lane), ak);
  }
#pragma unroll
  for (int r = 0; r < 8; ++r) {
    int m = (lane >> 4) * 8 + r, col = lane & 15;
    cqp[(n0 + m) * ATOM_Z + col] = aq[r];
    ckp[(n0 + m) * ATOM_Z + col] = ak[r];
  }
}

// ---------------------------------------------------------------------------
// Kernel 6: pair features + zp/cq/ck gathers + 3-layer 16x16 MLP (WMMA).
// One wave per 16-key tile of one (window, query). block 256; grid 4096.
// Key l of window kk is atom 32*kk - 48 + l (zero padded at edges).
// ---------------------------------------------------------------------------
__global__ __launch_bounds__(256) void k_p(
    const float* __restrict__ pos, const unsigned char* __restrict__ pmask,
    const int* __restrict__ uid, const int* __restrict__ tok,
    const float* __restrict__ cqp, const float* __restrict__ ckp,
    const float* __restrict__ zp, const float* __restrict__ Wpos,
    const float* __restrict__ Wdist, const float* __restrict__ Wmsk,
    const _Float16* __restrict__ hWm1, const _Float16* __restrict__ hWm2,
    const _Float16* __restrict__ hWm3, float* __restrict__ p_out) {
  __shared__ float P0[8][16 * LDP];
  __shared__ float Ht[8][16 * LDP];
  int tid = threadIdx.x, wave = tid >> 5, lane = tid & 31;

  int tileId = blockIdx.x * 8 + wave;
  int lt = tileId & 7;           // key sub-tile (16 keys)
  int w  = (tileId >> 3) & 31;   // query within window
  int kk = tileId >> 8;          // window

  int aq = kk * 32 + w;
  int tq = tok[aq];
  float pq0 = pos[aq * 3], pq1 = pos[aq * 3 + 1], pq2 = pos[aq * 3 + 2];
  int uq = uid[aq];
  bool mq = pmask[aq] != 0;

  // lane -> key row = lane&15; half of the 16 dz channels = (lane>>4)*8
  int li = lane & 15;
  int l  = lt * 16 + li;
  int ka = kk * 32 - 48 + l;
  bool inr = (ka >= 0) && (ka < N_ATOMS);
  float pk0 = 0.f, pk1 = 0.f, pk2 = 0.f;
  int ukv = -1, tkk = 0;
  bool mk = false;
  if (inr) {
    pk0 = pos[ka * 3]; pk1 = pos[ka * 3 + 1]; pk2 = pos[ka * 3 + 2];
    ukv = uid[ka];
    tkk = tok[ka];
    mk  = pmask[ka] != 0;
  }
  float d0 = pk0 - pq0, d1 = pk1 - pq1, d2 = pk2 - pq2;
  float dn = 1.f / (1.f + d0 * d0 + d1 * d1 + d2 * d2);
  float vm = (mq && mk && (uq == ukv)) ? 1.f : 0.f;

  int dz0 = (lane >> 4) * 8;
  float* prow = &P0[wave][li * LDP];
  const float* zrow = zp + ((long)tq * N_TOK + tkk) * ATOM_Z;
#pragma unroll
  for (int j = 0; j < 8; ++j) {
    int dz = dz0 + j;
    float val = vm * (d0 * Wpos[dz] + d1 * Wpos[16 + dz] + d2 * Wpos[32 + dz] +
                      dn * Wdist[dz] + Wmsk[dz]);   // (v@Wmask)*v == v*Wmask
    val += cqp[aq * ATOM_Z + dz];
    if (inr) val += ckp[ka * ATOM_Z + dz] + zrow[dz];
    prow[dz] = val;
  }
  __syncthreads();

  // h1 = relu(p0) @ Wm1
  v8f h = wmma_f16(afrag16_relu(P0[wave], lane), bload(hWm1, 0, lane), (v8f){});
#pragma unroll
  for (int r = 0; r < 8; ++r)
    Ht[wave][((lane >> 4) * 8 + r) * LDP + (lane & 15)] = h[r];
  __syncthreads();

  // h2 = relu(h1) @ Wm2
  h = wmma_f16(afrag16_relu(Ht[wave], lane), bload(hWm2, 0, lane), (v8f){});
  __syncthreads();
#pragma unroll
  for (int r = 0; r < 8; ++r)
    Ht[wave][((lane >> 4) * 8 + r) * LDP + (lane & 15)] = h[r];
  __syncthreads();

  // p = p0 + relu(h2) @ Wm3
  h = wmma_f16(afrag16_relu(Ht[wave], lane), bload(hWm3, 0, lane), (v8f){});
#pragma unroll
  for (int r = 0; r < 8; ++r) {
    int m = (lane >> 4) * 8 + r, col = lane & 15;
    float pv = P0[wave][m * LDP + col] + h[r];
    p_out[(((long)kk * 32 + w) * 128 + lt * 16 + m) * ATOM_Z + col] = pv;
  }
}

// ---------------------------------------------------------------------------
// Launch
// ---------------------------------------------------------------------------
extern "C" void kernel_launch(void* const* d_in, const int* in_sizes, int n_in,
                              void* d_out, int out_size, void* d_ws,
                              size_t ws_size, hipStream_t stream) {
  (void)in_sizes; (void)n_in; (void)out_size; (void)ws_size;

  const float*         ref_pos   = (const float*)d_in[0];
  const float*         ref_chg   = (const float*)d_in[1];
  const float*         ref_elem  = (const float*)d_in[2];
  const float*         ref_chars = (const float*)d_in[3];
  const unsigned char* pad_mask  = (const unsigned char*)d_in[4];
  const int*           uid       = (const int*)d_in[5];
  const float*         a2t       = (const float*)d_in[6];
  const float*         s_trunk   = (const float*)d_in[7];
  const float*         z         = (const float*)d_in[8];
  const float*         W_feat    = (const float*)d_in[9];
  const float*         b_feat    = (const float*)d_in[10];
  const float*         W_pos     = (const float*)d_in[11];
  const float*         W_dist    = (const float*)d_in[12];
  const float*         W_mask    = (const float*)d_in[13];
  const float*         ln_s_g    = (const float*)d_in[14];
  const float*         ln_s_b    = (const float*)d_in[15];
  const float*         W_s2c     = (const float*)d_in[16];
  const float*         ln_z_g    = (const float*)d_in[17];
  const float*         ln_z_b    = (const float*)d_in[18];
  const float*         W_z2p     = (const float*)d_in[19];
  const float*         W_cq      = (const float*)d_in[20];
  const float*         W_ck      = (const float*)d_in[21];
  const float*         W_m1      = (const float*)d_in[22];
  const float*         W_m2      = (const float*)d_in[23];
  const float*         W_m3      = (const float*)d_in[24];

  // f32 workspace
  float* ws  = (float*)d_ws;
  float* c0  = ws;                     // 4096*128
  float* c1  = c0 + 524288;            // 4096*128
  float* s2c = c1 + 524288;            // 512*128
  float* cqp = s2c + 65536;            // 4096*16
  float* ckp = cqp + 65536;            // 4096*16
  float* zpb = ckp + 65536;            // 512*512*16
  int*   tok = (int*)(zpb + 4194304);  // 4096
  // f16 swizzled-weight workspace (32B-aligned: offset 5443584 f32)
  _Float16* hbase = (_Float16*)(ws + 5443584);
  _Float16* hWfeat = hbase;            // 13*8*512 = 53248
  _Float16* hWs2c  = hWfeat + 53248;   // 12*8*512 = 49152
  _Float16* hWz2p  = hWs2c + 49152;    // 4*512
  _Float16* hWcq   = hWz2p + 2048;     // 4*512
  _Float16* hWck   = hWcq + 2048;      // 4*512
  _Float16* hWm1   = hWck + 2048;      // 512
  _Float16* hWm2   = hWm1 + 512;
  _Float16* hWm3   = hWm2 + 512;

  float* q_out = (float*)d_out;        // 4096*128
  float* c_out = q_out + 524288;       // 4096*128
  float* p_out = c_out + 524288;       // 128*32*128*16

  // weight swizzle (B-fragment order, f16, zero-padded)
  k_swzB<<<208, 256, 0, stream>>>(W_feat, FEATN, ATOM_S, 104, 8, hWfeat);
  k_swzB<<<192, 256, 0, stream>>>(W_s2c, TOKEN_S, ATOM_S, 96, 8, hWs2c);
  k_swzB<<<8,   256, 0, stream>>>(W_z2p, TOKEN_Z, ATOM_Z, 4, 1, hWz2p);
  k_swzB<<<8,   256, 0, stream>>>(W_cq,  TOKEN_Z, ATOM_Z, 4, 1, hWcq);
  k_swzB<<<8,   256, 0, stream>>>(W_ck,  TOKEN_Z, ATOM_Z, 4, 1, hWck);
  k_swzB<<<2,   256, 0, stream>>>(W_m1, 16, 16, 1, 1, hWm1);
  k_swzB<<<2,   256, 0, stream>>>(W_m2, 16, 16, 1, 1, hWm2);
  k_swzB<<<2,   256, 0, stream>>>(W_m3, 16, 16, 1, 1, hWm3);

  k_embed<<<256, 256, 0, stream>>>(ref_pos, ref_chg, ref_elem, ref_chars,
                                   hWfeat, b_feat, c0, q_out);
  k_s2c<<<32, 256, 0, stream>>>(s_trunk, ln_s_g, ln_s_b, hWs2c, s2c);
  k_zp<<<4096, 128, 0, stream>>>(z, ln_z_g, ln_z_b, hWz2p, zpb);
  k_token<<<512, 256, 0, stream>>>(a2t, c0, s2c, c1, c_out, tok);
  k_proj<<<32, 256, 0, stream>>>(c1, hWcq, hWck, cqp, ckp);
  k_p<<<4096, 256, 0, stream>>>(ref_pos, pad_mask, uid, tok, cqp, ckp, zpb,
                                W_pos, W_dist, W_mask, hWm1, hWm2, hWm3, p_out);
}
